// Net_90417651516003
// MI455X (gfx1250) — matrix operand order repaired and checked by reading, not compiled, not run on hardware
//
#include <hip/hip_runtime.h>

typedef _Float16 v16h __attribute__((ext_vector_type(16)));
typedef _Float16 v8h  __attribute__((ext_vector_type(8)));
typedef float    v8f  __attribute__((ext_vector_type(8)));
typedef float    f4   __attribute__((ext_vector_type(4)));
typedef int      v8i  __attribute__((ext_vector_type(8)));

#define T_STEPS 512
#define D_IN    128
#define HID     64
#define N_OUT   20

// ---------------- helpers ----------------

__device__ __forceinline__ v8f wmma_f16(v16h a, v16h b, v8f c) {
  // D = A(16x32 f16) x B(32x16 f16) + C(16x16 f32)
  return __builtin_amdgcn_wmma_f32_16x16x32_f16(false, a, false, b, (short)0, c,
                                                false, false);
}

__device__ __forceinline__ int pk2(float a, float b) {
  auto p = __builtin_amdgcn_cvt_pkrtz(a, b);   // packed f16 pair
  return __builtin_bit_cast(int, p);
}

__device__ __forceinline__ int swz16(int v) {
  // ds_swizzle SWAPX16: group-of-32, xor_mask=0x10, and_mask=0x1f
  return __builtin_amdgcn_ds_swizzle(v, 0x401f);
}

// Load an f16 A-fragment (16x32 tile, this lane's 16 halves):
// halves 0..15 = row[k0 .. k0+15]
__device__ __forceinline__ v16h load_afrag(const _Float16* p) {
  v8h a = *(const v8h*)p;
  v8h b = *(const v8h*)(p + 8);
  return __builtin_shufflevector(a, b, 0,1,2,3,4,5,6,7,8,9,10,11,12,13,14,15);
}

// Build an f16 B-fragment (32x16) for this lane from 16 consecutive f32 (streamed, NT)
__device__ __forceinline__ v16h load_xbfrag(const float* p) {
  f4 a = __builtin_nontemporal_load((const f4*)(p + 0));
  f4 b = __builtin_nontemporal_load((const f4*)(p + 4));
  f4 c = __builtin_nontemporal_load((const f4*)(p + 8));
  f4 d = __builtin_nontemporal_load((const f4*)(p + 12));
  v8i w;
  w[0] = pk2(a[0], a[1]); w[1] = pk2(a[2], a[3]);
  w[2] = pk2(b[0], b[1]); w[3] = pk2(b[2], b[3]);
  w[4] = pk2(c[0], c[1]); w[5] = pk2(c[2], c[3]);
  w[6] = pk2(d[0], d[1]); w[7] = pk2(d[2], d[3]);
  return __builtin_bit_cast(v16h, w);
}

// In-register transpose of spike C/D tiles into a B-fragment for the next layer.
// lo = spike tile (units 2kc*16..), hi = tile (2kc*16+16..); g = lane group (lane>>4).
// Cross-lane-group exchange done with ds_swizzle SWAPX16 on packed f16 pairs.
__device__ __forceinline__ v16h spikes_to_bfrag(v8f lo, v8f hi, int g) {
  v8i w;
  #pragma unroll
  for (int r2 = 0; r2 < 4; ++r2) {
    int l  = pk2(lo[2*r2], lo[2*r2 + 1]);
    int h  = pk2(hi[2*r2], hi[2*r2 + 1]);
    int ls = swz16(l);
    int hs = swz16(h);
    w[r2]     = g ? hs : l;   // halves 2r2, 2r2+1   (K within chunk: 16g + h, h<8)
    w[r2 + 4] = g ? h  : ls;  // halves 2r2+8,2r2+9  (K within chunk: 16g + h, h>=8)
  }
  return __builtin_bit_cast(v16h, w);
}

__device__ __forceinline__ v8f load8f(const float* p) {
  f4 a = *(const f4*)p;
  f4 b = *(const f4*)(p + 4);
  v8f r;
  r[0]=a[0]; r[1]=a[1]; r[2]=a[2]; r[3]=a[3];
  r[4]=b[0]; r[5]=b[1]; r[6]=b[2]; r[7]=b[3];
  return r;
}

__device__ __forceinline__ v8f clip01(v8f v) {
  v8f r;
  #pragma unroll
  for (int e = 0; e < 8; ++e) {
    float c = v[e];
    r[e] = c < 0.f ? 0.f : (c > 1.f ? 1.f : c);
  }
  return r;
}

// ---------------- prep: f32 weights -> f16 (W3 zero-padded 20->32 rows) ----------------

__global__ void prep_weights(const float* __restrict__ W1, const float* __restrict__ W2,
                             const float* __restrict__ W3, _Float16* __restrict__ wh) {
  int i = blockIdx.x * 256 + threadIdx.x;
  if (i < HID * D_IN) {                         // W1: 64x128
    wh[i] = (_Float16)W1[i];
  } else if (i < HID * D_IN + HID * HID) {      // W2: 64x64
    wh[i] = (_Float16)W2[i - HID * D_IN];
  } else if (i < HID * D_IN + HID * HID + 32 * HID) {  // W3 padded: 32x64
    int k = i - (HID * D_IN + HID * HID);
    int row = k >> 6, col = k & 63;
    wh[i] = (row < N_OUT) ? (_Float16)W3[row * HID + col] : (_Float16)0.f;
  }
}

// ---------------- main SNN kernel: one wave per 16-row batch tile ----------------

__global__ __launch_bounds__(32) void snn_forward(
    const float* __restrict__ x,
    const float* __restrict__ b1p, const float* __restrict__ bt1p,
    const float* __restrict__ b2p, const float* __restrict__ bt2p,
    const float* __restrict__ b3p, const float* __restrict__ bt3p,
    const _Float16* __restrict__ wh,
    float* __restrict__ out)
{
  const int lane = threadIdx.x & 31;
  const int n = lane & 15;     // batch column within tile / A-row within tile
  const int g = lane >> 4;     // lane group

  const _Float16* W1h = wh;
  const _Float16* W2h = wh + HID * D_IN;
  const _Float16* W3h = wh + HID * D_IN + HID * HID;

  // Weight A-fragments, resident in VGPRs across the whole T loop.
  v16h w1f[4][4];   // hidden tiles x K-chunks of 32 over D_IN=128
  #pragma unroll
  for (int mt = 0; mt < 4; ++mt)
    #pragma unroll
    for (int kc = 0; kc < 4; ++kc)
      w1f[mt][kc] = load_afrag(W1h + (mt*16 + n) * D_IN + kc*32 + 16*g);

  v16h w2f[4][2];   // hidden tiles x K-chunks over HID=64
  #pragma unroll
  for (int mt = 0; mt < 4; ++mt)
    #pragma unroll
    for (int kc = 0; kc < 2; ++kc)
      w2f[mt][kc] = load_afrag(W2h + (mt*16 + n) * HID + kc*32 + 16*g);

  v16h w3f[2][2];   // output tiles (padded 32) x K-chunks over HID=64
  #pragma unroll
  for (int mt = 0; mt < 2; ++mt)
    #pragma unroll
    for (int kc = 0; kc < 2; ++kc)
      w3f[mt][kc] = load_afrag(W3h + (mt*16 + n) * HID + kc*32 + 16*g);

  // Bias / clipped-beta in C-layout: element e of tile mt maps to unit mt*16 + 8g + e
  v8f b1v[4], bt1v[4], b2v[4], bt2v[4], b3v[2], bt3v[2];
  #pragma unroll
  for (int mt = 0; mt < 4; ++mt) {
    b1v[mt]  = load8f(b1p  + mt*16 + 8*g);
    bt1v[mt] = clip01(load8f(bt1p + mt*16 + 8*g));
    b2v[mt]  = load8f(b2p  + mt*16 + 8*g);
    bt2v[mt] = clip01(load8f(bt2p + mt*16 + 8*g));
  }
  #pragma unroll
  for (int mt = 0; mt < 2; ++mt) {
    v8f bb = {}, tt = {};
    #pragma unroll
    for (int e = 0; e < 8; ++e) {
      int o = mt*16 + 8*g + e;
      if (o < N_OUT) {
        bb[e] = b3p[o];
        float c = bt3p[o];
        tt[e] = c < 0.f ? 0.f : (c > 1.f ? 1.f : c);
      }
    }
    b3v[mt] = bb; bt3v[mt] = tt;
  }

  // Membrane state (transposed C-layout), resident for all 512 steps.
  v8f m1[4] = {}, m2[4] = {}, m3v[2] = {};

  const long rb = (long)blockIdx.x * 16 + n;      // this lane's batch row
  const float* xp = x + rb * T_STEPS * D_IN + 16 * g;
  float* op = out + rb * T_STEPS * N_OUT;

  #pragma unroll 1
  for (int t = 0; t < T_STEPS; ++t) {
    const float* xt = xp + t * D_IN;
    if (t + 4 < T_STEPS)
      __builtin_prefetch(xt + 4 * D_IN, 0, 0);    // global_prefetch a few steps ahead

    // ---- layer 1: cur1^T = W1 @ x_t^T ----
    v16h xb[4];
    #pragma unroll
    for (int kc = 0; kc < 4; ++kc) xb[kc] = load_xbfrag(xt + kc * 32);

    v8f s1[4];
    #pragma unroll
    for (int mt = 0; mt < 4; ++mt) {
      v8f acc = b1v[mt];
      #pragma unroll
      for (int kc = 0; kc < 4; ++kc) acc = wmma_f16(w1f[mt][kc], xb[kc], acc);
      v8f mo = m1[mt], bt = bt1v[mt], mn, sp;
      #pragma unroll
      for (int e = 0; e < 8; ++e) {
        float r = mo[e] > 1.0f ? 1.0f : 0.0f;               // reset from old mem
        float v = __builtin_fmaf(bt[e], mo[e], acc[e]) - r; // beta*m + cur - r*TH
        mn[e] = v;
        sp[e] = v > 1.0f ? 1.0f : 0.0f;                     // spike(m - TH)
      }
      m1[mt] = mn; s1[mt] = sp;
    }

    // ---- layer 2: cur2^T = W2 @ s1^T ----
    v16h s1b[2];
    #pragma unroll
    for (int kc = 0; kc < 2; ++kc)
      s1b[kc] = spikes_to_bfrag(s1[2*kc], s1[2*kc + 1], g);

    v8f s2[4];
    #pragma unroll
    for (int mt = 0; mt < 4; ++mt) {
      v8f acc = b2v[mt];
      #pragma unroll
      for (int kc = 0; kc < 2; ++kc) acc = wmma_f16(w2f[mt][kc], s1b[kc], acc);
      v8f mo = m2[mt], bt = bt2v[mt], mn, sp;
      #pragma unroll
      for (int e = 0; e < 8; ++e) {
        float r = mo[e] > 1.0f ? 1.0f : 0.0f;
        float v = __builtin_fmaf(bt[e], mo[e], acc[e]) - r;
        mn[e] = v;
        sp[e] = v > 1.0f ? 1.0f : 0.0f;
      }
      m2[mt] = mn; s2[mt] = sp;
    }

    // ---- layer 3: m3 = beta3*m3 + W3 @ s2^T + b3 (no reset) ----
    v16h s2b[2];
    #pragma unroll
    for (int kc = 0; kc < 2; ++kc)
      s2b[kc] = spikes_to_bfrag(s2[2*kc], s2[2*kc + 1], g);

    #pragma unroll
    for (int mt = 0; mt < 2; ++mt) {
      v8f acc = b3v[mt];
      #pragma unroll
      for (int kc = 0; kc < 2; ++kc) acc = wmma_f16(w3f[mt][kc], s2b[kc], acc);
      v8f mo = m3v[mt], bt = bt3v[mt], mn;
      #pragma unroll
      for (int e = 0; e < 8; ++e) mn[e] = __builtin_fmaf(bt[e], mo[e], acc[e]);
      m3v[mt] = mn;
    }

    // ---- store V[b][t][0..19]; element e of tile mt is unit mt*16 + 8g + e ----
    float* o = op + t * N_OUT;
    f4 q0 = { m3v[0][0], m3v[0][1], m3v[0][2], m3v[0][3] };
    f4 q1 = { m3v[0][4], m3v[0][5], m3v[0][6], m3v[0][7] };
    if (g == 0) {
      __builtin_nontemporal_store(q0, (f4*)(o + 0));
      __builtin_nontemporal_store(q1, (f4*)(o + 4));
      f4 q2 = { m3v[1][0], m3v[1][1], m3v[1][2], m3v[1][3] };
      __builtin_nontemporal_store(q2, (f4*)(o + 16));
    } else {
      __builtin_nontemporal_store(q0, (f4*)(o + 8));
      __builtin_nontemporal_store(q1, (f4*)(o + 12));
    }
  }
}

// ---------------- launch ----------------

extern "C" void kernel_launch(void* const* d_in, const int* in_sizes, int n_in,
                              void* d_out, int out_size, void* d_ws, size_t ws_size,
                              hipStream_t stream) {
  const float* x     = (const float*)d_in[0];
  const float* W1    = (const float*)d_in[1];
  const float* b1    = (const float*)d_in[2];
  const float* beta1 = (const float*)d_in[3];
  const float* W2    = (const float*)d_in[4];
  const float* b2    = (const float*)d_in[5];
  const float* beta2 = (const float*)d_in[6];
  const float* W3    = (const float*)d_in[7];
  const float* b3    = (const float*)d_in[8];
  const float* beta3 = (const float*)d_in[9];
  float* out = (float*)d_out;
  _Float16* wh = (_Float16*)d_ws;   // 64*128 + 64*64 + 32*64 = 14336 halves = 28 KB

  const int nhalf = HID * D_IN + HID * HID + 32 * HID;
  prep_weights<<<(nhalf + 255) / 256, 256, 0, stream>>>(W1, W2, W3, wh);

  const int B = in_sizes[0] / (T_STEPS * D_IN);   // 1024
  snn_forward<<<B / 16, 32, 0, stream>>>(x, b1, beta1, b2, beta2, b3, beta3, wh, out);
}
